// BlockGTrXL_5437428597401
// MI455X (gfx1250) — matrix operand order repaired and checked by reading, not compiled, not run on hardware
//
#include <hip/hip_runtime.h>
#include <math.h>

// ---------------------------------------------------------------------------
// GTrXL block for MI455X (gfx1250): bf16 WMMA GEMMs + flash attention.
// All GEMMs: V_WMMA_F32_16X16X32_BF16, f32 accumulate, wave32.
// Round 2: transposed-B LDS layout (b128 fragment reads instead of 16x
// ds_load_u16), 64x128 GEMM tile (4 WMMA per wave per K-step), coalesced
// V staging through LDS in the attention kernel.
// ---------------------------------------------------------------------------

typedef __attribute__((ext_vector_type(16))) __bf16 v16bf;
typedef __attribute__((ext_vector_type(8)))  float  v8f;

struct alignas(16) BF8 { __bf16 v[8]; };
struct alignas(16) F4  { float  v[4]; };

#define T_SEQ 1024
#define CH    1024
#define NHEAD 16
#define DH    64

__device__ __forceinline__ v8f wmma_bf16(v16bf a, v16bf b, v8f c) {
  // D = A(16x32 bf16) * B(32x16 bf16) + C(16x16 f32)
  return __builtin_amdgcn_wmma_f32_16x16x32_bf16(false, a, false, b, (short)0, c,
                                                 false, false);
}

// A-fragment (16x32, 16-bit): lane row = lane&15; K elems: half*8+j, 16+half*8+j
__device__ __forceinline__ v16bf frag_a_lds(const __bf16* base, int ldm, int lane) {
  const int half = (lane >> 4) & 1;
  const __bf16* p = base + (size_t)(lane & 15) * ldm + half * 8;
  v16bf a;
#pragma unroll
  for (int j = 0; j < 8; ++j) a[j] = p[j];
#pragma unroll
  for (int j = 0; j < 8; ++j) a[j + 8] = p[16 + j];
  return a;
}

// B-fragment (32x16, 16-bit) from a TRANSPOSED [n][k] LDS tile:
// lane col = lane&15, K elems: half*8+j, 16+half*8+j (two 8-elem runs per lane).
__device__ __forceinline__ v16bf frag_b_ldsT(const __bf16* Bst, int ldk, int col,
                                             int lane) {
  const int half = (lane >> 4) & 1;
  const __bf16* p = Bst + (size_t)(col + (lane & 15)) * ldk + half * 8;
  v16bf b;
#pragma unroll
  for (int j = 0; j < 8; ++j) { b[j] = p[j]; b[j + 8] = p[16 + j]; }
  return b;
}

// ---------------------------------------------------------------------------
// GEMM: out = A(bf16, MxK) * B(f32 weights, KxN, converted to bf16 on load)
//             [+ Cacc] [+ bias_scale * bias] ; write f32 and/or bf16.
// 256 threads = 8 waves; 64x128 tile; K-step 32; per wave: 16x64 (4 WMMAs).
// B staged TRANSPOSED in LDS so fragments read contiguously (ds_load_b128).
// ---------------------------------------------------------------------------
__global__ __launch_bounds__(256)
void gemm_bf16f32_kernel(const __bf16* __restrict__ A, const float* __restrict__ B,
                         const float* __restrict__ bias, float bias_scale,
                         const float* Cacc, float* outF, __bf16* outB,
                         int M, int N, int K)
{
  __shared__ __bf16 As[64][48];     // 64x32 used; stride 48 keeps 16B alignment
  __shared__ __bf16 Bst[128][40];   // transposed B tile [n][k]; stride 40 (80B)

  const int tid   = threadIdx.x;
  const int lane  = tid & 31;
  const int wave  = tid >> 5;
  const int m0    = blockIdx.y * 64;
  const int n0    = blockIdx.x * 128;
  const int m_off = (wave & 3) * 16;
  const int n_off = (wave >> 2) * 64;

  const int ar = tid >> 2,       ac = (tid & 3) * 8;   // A tile: 64 rows x 32 cols
  const int kk = (tid >> 4) * 2;                       // even k-row pair (0..30)
  const int cg = (tid & 15) * 8;                       // col group (0..120)

  const __bf16* Aload = A + (size_t)(m0 + ar) * K + ac;
  const float*  B0 = B + (size_t)kk * N + n0 + cg;     // row kk
  const float*  B1 = B0 + N;                           // row kk+1

  v8f acc0 = {}, acc1 = {}, acc2 = {}, acc3 = {};

  for (int k0 = 0; k0 < K; k0 += 32) {
    // stage A (row-major) and B (transposed, f32 -> bf16 pairs, b32 stores)
    *(BF8*)&As[ar][ac] = *(const BF8*)(Aload + k0);
    F4 r0a = *(const F4*)(B0 + (size_t)k0 * N);
    F4 r0b = *(const F4*)(B0 + (size_t)k0 * N + 4);
    F4 r1a = *(const F4*)(B1 + (size_t)k0 * N);
    F4 r1b = *(const F4*)(B1 + (size_t)k0 * N + 4);
#pragma unroll
    for (int j = 0; j < 4; ++j) {
      union { __bf16 h[2]; unsigned u; } pk;
      pk.h[0] = (__bf16)r0a.v[j]; pk.h[1] = (__bf16)r1a.v[j];
      *(unsigned*)&Bst[cg + j][kk] = pk.u;
      pk.h[0] = (__bf16)r0b.v[j]; pk.h[1] = (__bf16)r1b.v[j];
      *(unsigned*)&Bst[cg + 4 + j][kk] = pk.u;
    }
    if (k0 + 32 < K) {
      __builtin_prefetch(Aload + k0 + 32, 0, 1);
      __builtin_prefetch(B0 + (size_t)(k0 + 32) * N, 0, 1);
    }
    __syncthreads();
    v16bf a  = frag_a_lds(&As[m_off][0], 48, lane);
    v16bf b0 = frag_b_ldsT(&Bst[0][0], 40, n_off + 0,  lane);
    v16bf b1 = frag_b_ldsT(&Bst[0][0], 40, n_off + 16, lane);
    v16bf b2 = frag_b_ldsT(&Bst[0][0], 40, n_off + 32, lane);
    v16bf b3 = frag_b_ldsT(&Bst[0][0], 40, n_off + 48, lane);
    acc0 = wmma_bf16(a, b0, acc0);
    acc1 = wmma_bf16(a, b1, acc1);
    acc2 = wmma_bf16(a, b2, acc2);
    acc3 = wmma_bf16(a, b3, acc3);
    __syncthreads();
  }

  // C/D layout: lane<16 -> M=i, N=lane ; lane>=16 -> M=i+8, N=lane-16
  const int half = lane >> 4, cl = lane & 15;
#pragma unroll
  for (int i = 0; i < 8; ++i) {
    const int gr = m0 + m_off + half * 8 + i;
    float vv[4] = { acc0[i], acc1[i], acc2[i], acc3[i] };
#pragma unroll
    for (int s = 0; s < 4; ++s) {
      const int gc = n0 + n_off + s * 16 + cl;
      float val = vv[s];
      if (Cacc) val += Cacc[(size_t)gr * N + gc];
      if (bias) val += bias_scale * bias[gc];
      if (outF) outF[(size_t)gr * N + gc] = val;
      if (outB) outB[(size_t)gr * N + gc] = (__bf16)val;
    }
  }
}

// ---------------------------------------------------------------------------
// Flash attention: 1 wave per (b, h, 16-query tile). Causal. Writes raw
// masked scores to att (pre-filled with -inf), softmax(P)@V to ybf (bf16).
// V is staged through LDS with coalesced b128 loads (1 row per lane).
// ---------------------------------------------------------------------------
__global__ __launch_bounds__(32)
void attn_flash_kernel(const __bf16* __restrict__ qh, const __bf16* __restrict__ kh,
                       const __bf16* __restrict__ vh, float* __restrict__ att,
                       __bf16* __restrict__ ybf)
{
  const int qt = blockIdx.x, h = blockIdx.y, b = blockIdx.z;
  const int q0 = qt * 16;
  const int lane = threadIdx.x;
  const int half = lane >> 4, cl = lane & 15;
  __shared__ __bf16 sP[16][48];   // P tile 16x32 row-major
  __shared__ __bf16 sV[32][72];   // V tile 32(keys) x 64(d) row-major

  const size_t rowbase = (size_t)(b * T_SEQ) * CH + h * DH;

  // Q fragments: rows q0..q0+15, K(d)=0..63 -> two 16x32 A frags
  v16bf aQ0, aQ1;
  {
    const __bf16* p = qh + rowbase + (size_t)(q0 + cl) * CH + half * 8;
#pragma unroll
    for (int j = 0; j < 8; ++j) {
      aQ0[j] = p[j];      aQ0[8 + j] = p[16 + j];
      aQ1[j] = p[32 + j]; aQ1[8 + j] = p[48 + j];
    }
  }

  float mrow[8], lrow[8];
#pragma unroll
  for (int i = 0; i < 8; ++i) { mrow[i] = -__builtin_inff(); lrow[i] = 0.0f; }
  v8f o0 = {}, o1 = {}, o2 = {}, o3 = {};

  float* attb = att + ((size_t)(b * NHEAD + h)) * T_SEQ * T_SEQ;
  const int nch = (q0 + 16 + 31) >> 5;   // 32-key chunks covering keys <= q0+15

  for (int kc = 0; kc < nch; ++kc) {
    const int kb = kc * 32;
    // S = Q * K^T : K^T B-frag lane = key column, contiguous d per kh row
    v8f s0 = {}, s1 = {};
    {
      const __bf16* kp = kh + rowbase + (size_t)(kb + cl) * CH + half * 8;
      v16bf b0, b1;
#pragma unroll
      for (int j = 0; j < 8; ++j) { b0[j] = kp[j]; b0[8 + j] = kp[16 + j]; b1[j] = kp[32 + j]; b1[8 + j] = kp[48 + j]; }
      s0 = wmma_bf16(aQ0, b0, s0);
      s0 = wmma_bf16(aQ1, b1, s0);
      const __bf16* kp2 = kp + (size_t)16 * CH;
#pragma unroll
      for (int j = 0; j < 8; ++j) { b0[j] = kp2[j]; b0[8 + j] = kp2[16 + j]; b1[j] = kp2[32 + j]; b1[8 + j] = kp2[48 + j]; }
      s1 = wmma_bf16(aQ0, b0, s1);
      s1 = wmma_bf16(aQ1, b1, s1);
    }

    float resc[8];
#pragma unroll
    for (int i = 0; i < 8; ++i) {
      const int qr  = q0 + half * 8 + i;
      const int k0i = kb + cl, k1i = k0i + 16;
      float v0 = s0[i] * 0.125f; if (k0i > qr) v0 = -__builtin_inff();
      float v1 = s1[i] * 0.125f; if (k1i > qr) v1 = -__builtin_inff();
      attb[(size_t)qr * T_SEQ + k0i] = v0;
      attb[(size_t)qr * T_SEQ + k1i] = v1;
      float mx = fmaxf(v0, v1);
      mx = fmaxf(mx, __shfl_xor(mx, 1, 32));
      mx = fmaxf(mx, __shfl_xor(mx, 2, 32));
      mx = fmaxf(mx, __shfl_xor(mx, 4, 32));
      mx = fmaxf(mx, __shfl_xor(mx, 8, 32));
      const float mnew = fmaxf(mrow[i], mx);
      const float rs = __expf(mrow[i] - mnew);
      const float e0 = __expf(v0 - mnew), e1 = __expf(v1 - mnew);
      float sum = e0 + e1;
      sum += __shfl_xor(sum, 1, 32);
      sum += __shfl_xor(sum, 2, 32);
      sum += __shfl_xor(sum, 4, 32);
      sum += __shfl_xor(sum, 8, 32);
      lrow[i] = lrow[i] * rs + sum;
      mrow[i] = mnew;
      resc[i] = rs;
      sP[half * 8 + i][cl]      = (__bf16)e0;
      sP[half * 8 + i][16 + cl] = (__bf16)e1;
    }
#pragma unroll
    for (int i = 0; i < 8; ++i) {
      o0[i] *= resc[i]; o1[i] *= resc[i]; o2[i] *= resc[i]; o3[i] *= resc[i];
    }

    // stage V[kb..kb+31][0..31] cooperatively: 1 row per lane, 4 b128 loads
    {
      const __bf16* vr = vh + rowbase + (size_t)(kb + lane) * CH;
#pragma unroll
      for (int c = 0; c < 4; ++c)
        *(BF8*)&sV[lane][c * 8] = *(const BF8*)(vr + c * 8);
    }
    __syncthreads();

    v16bf aP;   // re-read P in A-fragment layout (contiguous per lane)
    {
      const __bf16* pp = &sP[cl][half * 8];
#pragma unroll
      for (int j = 0; j < 8; ++j) { aP[j] = pp[j]; aP[8 + j] = pp[16 + j]; }
    }
    // V B-frags from LDS: lane col = d, K = key-in-chunk
    v16bf bv0, bv1, bv2, bv3;
#pragma unroll
    for (int j = 0; j < 16; ++j) {
      const __bf16* vr = &sV[half * 8 + (j < 8 ? j : 8 + j)][0];
      bv0[j] = vr[cl]; bv1[j] = vr[16 + cl]; bv2[j] = vr[32 + cl]; bv3[j] = vr[48 + cl];
    }
    __syncthreads();

    o0 = wmma_bf16(aP, bv0, o0);
    o1 = wmma_bf16(aP, bv1, o1);
    o2 = wmma_bf16(aP, bv2, o2);
    o3 = wmma_bf16(aP, bv3, o3);
  }

#pragma unroll
  for (int i = 0; i < 8; ++i) {
    const float rl = 1.0f / lrow[i];
    __bf16* yr = ybf + rowbase + (size_t)(q0 + half * 8 + i) * CH;
    yr[cl]      = (__bf16)(o0[i] * rl);
    yr[16 + cl] = (__bf16)(o1[i] * rl);
    yr[32 + cl] = (__bf16)(o2[i] * rl);
    yr[48 + cl] = (__bf16)(o3[i] * rl);
  }
}

// ---------------------------------------------------------------------------
// LayerNorm (f32 in -> bf16 out), one row per 256-thread block.
// ---------------------------------------------------------------------------
__global__ __launch_bounds__(256)
void layernorm_bf16_kernel(const float* __restrict__ x, const float* __restrict__ g,
                           const float* __restrict__ bta, __bf16* __restrict__ out, int Cn)
{
  const int row = blockIdx.x;
  const int tid = threadIdx.x;
  const float* xr = x + (size_t)row * Cn;
  __shared__ float red[256];
  float s = 0.0f;
  for (int i = tid; i < Cn; i += 256) s += xr[i];
  red[tid] = s; __syncthreads();
  for (int o = 128; o > 0; o >>= 1) { if (tid < o) red[tid] += red[tid + o]; __syncthreads(); }
  const float mean = red[0] * (1.0f / Cn);
  __syncthreads();
  float vs = 0.0f;
  for (int i = tid; i < Cn; i += 256) { float d = xr[i] - mean; vs += d * d; }
  red[tid] = vs; __syncthreads();
  for (int o = 128; o > 0; o >>= 1) { if (tid < o) red[tid] += red[tid + o]; __syncthreads(); }
  const float rstd = rsqrtf(red[0] * (1.0f / Cn) + 1e-5f);
  __bf16* orow = out + (size_t)row * Cn;
  for (int i = tid; i < Cn; i += 256)
    orow[i] = (__bf16)((xr[i] - mean) * rstd * g[i] + bta[i]);
}

// ---------------------------------------------------------------------------
// Pointwise kernels
// ---------------------------------------------------------------------------
__global__ void fill_f32_kernel(float* p, float val, size_t n) {
  size_t i = (size_t)blockIdx.x * blockDim.x + threadIdx.x;
  const size_t st = (size_t)gridDim.x * blockDim.x;
  for (; i < n; i += st) p[i] = val;
}
__global__ void cvt_f32_bf16_kernel(const float* __restrict__ in, __bf16* __restrict__ out, size_t n) {
  size_t i = (size_t)blockIdx.x * blockDim.x + threadIdx.x;
  const size_t st = (size_t)gridDim.x * blockDim.x;
  for (; i < n; i += st) out[i] = (__bf16)in[i];
}
// r = sigmoid(gr); hr = bf16(h*r); gz <- sigmoid(gz)
__global__ void gru_rz_kernel(const float* __restrict__ gr, float* gz,
                              const float* __restrict__ h, __bf16* __restrict__ hr, size_t n) {
  size_t i = (size_t)blockIdx.x * blockDim.x + threadIdx.x;
  const size_t st = (size_t)gridDim.x * blockDim.x;
  for (; i < n; i += st) {
    const float r = 1.0f / (1.0f + __expf(-gr[i]));
    hr[i] = (__bf16)(h[i] * r);
    gz[i] = 1.0f / (1.0f + __expf(-gz[i]));
  }
}
// x = (1-z)*h + z*tanh(gh)
__global__ void gru_out_kernel(const float* __restrict__ gh, const float* __restrict__ z,
                               const float* __restrict__ h, float* __restrict__ xf,
                               __bf16* xb, size_t n) {
  size_t i = (size_t)blockIdx.x * blockDim.x + threadIdx.x;
  const size_t st = (size_t)gridDim.x * blockDim.x;
  for (; i < n; i += st) {
    const float zz = z[i];
    const float val = (1.0f - zz) * h[i] + zz * tanhf(gh[i]);
    xf[i] = val;
    if (xb) xb[i] = (__bf16)val;
  }
}
__global__ void gelu_bf16_kernel(const float* __restrict__ in, __bf16* __restrict__ out, size_t n) {
  size_t i = (size_t)blockIdx.x * blockDim.x + threadIdx.x;
  const size_t st = (size_t)gridDim.x * blockDim.x;
  for (; i < n; i += st) {
    const float x = in[i];
    out[i] = (__bf16)(0.5f * x * (1.0f + erff(x * 0.7071067811865476f)));
  }
}

// ---------------------------------------------------------------------------
extern "C" void kernel_launch(void* const* d_in, const int* in_sizes, int n_in,
                              void* d_out, int out_size, void* d_ws, size_t ws_size,
                              hipStream_t stream)
{
  (void)in_sizes; (void)n_in; (void)out_size; (void)ws_size;
  const float* q     = (const float*)d_in[0];
  const float* k     = (const float*)d_in[1];
  const float* v     = (const float*)d_in[2];
  const float* ln1_g = (const float*)d_in[3];
  const float* ln1_b = (const float*)d_in[4];
  const float* ln2_g = (const float*)d_in[5];
  const float* ln2_b = (const float*)d_in[6];
  const float* Wq = (const float*)d_in[7];  const float* bq = (const float*)d_in[8];
  const float* Wk = (const float*)d_in[9];  const float* bk = (const float*)d_in[10];
  const float* Wv = (const float*)d_in[11]; const float* bv = (const float*)d_in[12];
  const float* Wp = (const float*)d_in[13]; const float* bp = (const float*)d_in[14];
  const float* W1 = (const float*)d_in[15]; const float* b1 = (const float*)d_in[16];
  const float* W2 = (const float*)d_in[17]; const float* b2 = (const float*)d_in[18];
  const float* g1w  = (const float*)d_in[19];
  const float* g1u  = (const float*)d_in[20];
  const float* g1bz = (const float*)d_in[21];
  const float* g2w  = (const float*)d_in[22];
  const float* g2u  = (const float*)d_in[23];
  const float* g2bz = (const float*)d_in[24];

  const int Bb = 4, Cc = 1024, C2 = 2048, Mrows = 4096;
  const size_t NC  = (size_t)Mrows * Cc;          // 4 Mi elements
  const size_t CCs = (size_t)Cc * Cc;             // weight matrix stride

  float* xout = (float*)d_out;                    // x: [B,T,C] f32
  float* att  = (float*)d_out + NC;               // att_scores: [B,H,T,T] f32

  // ---- workspace carve (256B aligned) ----
  char* wsp = (char*)d_ws;
  size_t off = 0;
  auto alloc = [&](size_t bytes) -> void* {
    void* p = wsp + off; off += (bytes + 255) & ~(size_t)255; return p;
  };
  __bf16* ln_q = (__bf16*)alloc(NC * 2);   // later: y_bf (attention out)
  __bf16* ln_k = (__bf16*)alloc(NC * 2);   // later: Xb (gate1 X), Xb2 (gate2 X)
  __bf16* ln_v = (__bf16*)alloc(NC * 2);   // later: hr_bf (both gates)
  __bf16* qhb  = (__bf16*)alloc(NC * 2);   // later: h2_b, then gelu_b (spans qhb+khb)
  __bf16* khb  = (__bf16*)alloc(NC * 2);
  __bf16* vhb  = (__bf16*)alloc(NC * 2);   // later: x1_b
  __bf16* qb   = (__bf16*)alloc(NC * 2);   // raw q in bf16 (gate1 h)
  float*  t_r  = (float*)alloc(NC * 4);
  float*  t_z  = (float*)alloc(NC * 4);
  float*  t_h  = (float*)alloc(NC * 4);
  float*  x1f  = (float*)alloc(NC * 4);
  float*  tbig = (float*)alloc((size_t)Mrows * C2 * 4);

  __bf16* ybf   = ln_q;           // attention output (bf16)
  __bf16* Xb    = ln_k;           // att projection out (bf16)
  __bf16* hrb   = ln_v;           // h*r (bf16)
  __bf16* h2b   = qhb;            // LN2 out (bf16)
  __bf16* gelub = qhb;            // gelu out, 8 Mi elems spans qhb+khb
  __bf16* x1b   = vhb;            // gate1 out (bf16)
  __bf16* Xb2   = ln_k;           // mlp out (bf16)

  auto gemm = [&](const __bf16* A, const float* Bw, const float* bias, float bsc,
                  const float* Cacc, float* oF, __bf16* oB, int M_, int N_, int K_) {
    gemm_bf16f32_kernel<<<dim3(N_ / 128, M_ / 64), 256, 0, stream>>>(
        A, Bw, bias, bsc, Cacc, oF, oB, M_, N_, K_);
  };

  // 1) pre-fill att_scores with -inf (upper triangle / untouched chunks)
  fill_f32_kernel<<<8192, 256, 0, stream>>>(att, -__builtin_inff(),
                                            (size_t)Bb * NHEAD * T_SEQ * T_SEQ);
  // 2) q raw -> bf16 (GRU gate-1 hidden input)
  cvt_f32_bf16_kernel<<<4096, 256, 0, stream>>>(q, qb, NC);
  // 3) LN1 on q, k, v
  layernorm_bf16_kernel<<<Mrows, 256, 0, stream>>>(q, ln1_g, ln1_b, ln_q, Cc);
  layernorm_bf16_kernel<<<Mrows, 256, 0, stream>>>(k, ln1_g, ln1_b, ln_k, Cc);
  layernorm_bf16_kernel<<<Mrows, 256, 0, stream>>>(v, ln1_g, ln1_b, ln_v, Cc);
  // 4) Q/K/V projections -> bf16 heads
  gemm(ln_q, Wq, bq, 1.0f, nullptr, nullptr, qhb, Mrows, Cc, Cc);
  gemm(ln_k, Wk, bk, 1.0f, nullptr, nullptr, khb, Mrows, Cc, Cc);
  gemm(ln_v, Wv, bv, 1.0f, nullptr, nullptr, vhb, Mrows, Cc, Cc);
  // 5) causal flash attention (+ att_scores output)
  attn_flash_kernel<<<dim3(T_SEQ / 16, NHEAD, Bb), 32, 0, stream>>>(qhb, khb, vhb, att, ybf);
  // 6) output projection -> X (bf16) for gate 1
  gemm(ybf, Wp, bp, 1.0f, nullptr, nullptr, Xb, Mrows, Cc, Cc);
  // 7) GRU gate 1 (h = q)
  gemm(Xb, g1w,           nullptr, 0.0f, nullptr, t_r, nullptr, Mrows, Cc, Cc);
  gemm(qb, g1u,           nullptr, 0.0f, t_r,     t_r, nullptr, Mrows, Cc, Cc);
  gemm(Xb, g1w + CCs,     nullptr, 0.0f, nullptr, t_z, nullptr, Mrows, Cc, Cc);
  gemm(qb, g1u + CCs,     g1bz,   -1.0f, t_z,     t_z, nullptr, Mrows, Cc, Cc);
  gru_rz_kernel<<<4096, 256, 0, stream>>>(t_r, t_z, q, hrb, NC);
  gemm(Xb,  g1w + 2 * CCs, nullptr, 0.0f, nullptr, t_h, nullptr, Mrows, Cc, Cc);
  gemm(hrb, g1u + 2 * CCs, nullptr, 0.0f, t_h,     t_h, nullptr, Mrows, Cc, Cc);
  gru_out_kernel<<<4096, 256, 0, stream>>>(t_h, t_z, q, x1f, x1b, NC);
  // 8) LN2 + MLP (GELU exact)
  layernorm_bf16_kernel<<<Mrows, 256, 0, stream>>>(x1f, ln2_g, ln2_b, h2b, Cc);
  gemm(h2b, W1, b1, 1.0f, nullptr, tbig, nullptr, Mrows, C2, Cc);
  gelu_bf16_kernel<<<8192, 256, 0, stream>>>(tbig, gelub, (size_t)Mrows * C2);
  gemm(gelub, W2, b2, 1.0f, nullptr, nullptr, Xb2, Mrows, Cc, C2);
  // 9) GRU gate 2 (h = x1) -> final x into d_out
  gemm(Xb2, g2w,           nullptr, 0.0f, nullptr, t_r, nullptr, Mrows, Cc, Cc);
  gemm(x1b, g2u,           nullptr, 0.0f, t_r,     t_r, nullptr, Mrows, Cc, Cc);
  gemm(Xb2, g2w + CCs,     nullptr, 0.0f, nullptr, t_z, nullptr, Mrows, Cc, Cc);
  gemm(x1b, g2u + CCs,     g2bz,   -1.0f, t_z,     t_z, nullptr, Mrows, Cc, Cc);
  gru_rz_kernel<<<4096, 256, 0, stream>>>(t_r, t_z, x1f, hrb, NC);
  gemm(Xb2, g2w + 2 * CCs, nullptr, 0.0f, nullptr, t_h, nullptr, Mrows, Cc, Cc);
  gemm(hrb, g2u + 2 * CCs, nullptr, 0.0f, t_h,     t_h, nullptr, Mrows, Cc, Cc);
  gru_out_kernel<<<4096, 256, 0, stream>>>(t_h, t_z, x1f, xout, nullptr, NC);
}
